// DSConvBlockMMCV_35699768165209
// MI455X (gfx1250) — matrix operand order repaired and checked by reading, not compiled, not run on hardware
//
#include <hip/hip_runtime.h>
#include <hip/hip_bf16.h>

typedef __attribute__((ext_vector_type(16))) _Float16 v16h;
typedef __attribute__((ext_vector_type(8)))  _Float16 v8h;
typedef __attribute__((ext_vector_type(8)))  float    v8f;
typedef __attribute__((ext_vector_type(4)))  int      v4i;

#define BATCH 4
#define CH    64
#define HH    128
#define WW    128
#define HW    (HH*WW)        // 16384
#define KTOT  (CH*9)         // 576
#define NTOT  (BATCH*HW)     // 65536
#define GN_EPS 1e-5f

// Async global->LDS copy path (gfx1250): guard so the file compiles on any toolchain.
#if defined(__has_builtin)
#  if __has_builtin(__builtin_amdgcn_global_load_async_to_lds_b128) && \
      __has_builtin(__builtin_amdgcn_s_wait_asynccnt)
#    define HAVE_ASYNC_LDS 1
#  endif
#endif
#ifndef HAVE_ASYNC_LDS
#  define HAVE_ASYNC_LDS 0
#endif

#if HAVE_ASYNC_LDS
// Builtin signature (from hipcc diagnostic): param0 is v4i in AS(1) ("__device__"),
// param1 is the LDS-side v4i pointer, then two integer immediates (offset, cpol).
typedef __attribute__((address_space(1))) v4i g_v4i;
typedef __attribute__((address_space(3))) v4i l_v4i;
#endif

// ---------------- offset head: 3x3 conv (Cin -> 3) + activations ----------------
__global__ __launch_bounds__(256) void head_kernel(
    const float* __restrict__ src, const float* __restrict__ w,
    const float* __restrict__ bias, float* __restrict__ head)
{
  int tid = blockIdx.x * 256 + threadIdx.x;
  if (tid >= BATCH * HW) return;
  int b = tid / HW, p = tid % HW;
  int y = p / WW, x = p % WW;
  float a0 = bias[0], a1 = bias[1], a2 = bias[2];
  const float* xb = src + (size_t)b * CH * HW;
  for (int c = 0; c < CH; ++c) {
    const float* xc = xb + (size_t)c * HW;
    const float* w0 = w + (0 * CH + c) * 9;
    const float* w1 = w + (1 * CH + c) * 9;
    const float* w2 = w + (2 * CH + c) * 9;
    #pragma unroll
    for (int t = 0; t < 9; ++t) {
      int yy = y + t / 3 - 1, xx = x + t % 3 - 1;
      if (yy < 0 || yy >= HH || xx < 0 || xx >= WW) continue;
      float v = xc[yy * WW + xx];
      a0 = fmaf(v, w0[t], a0);
      a1 = fmaf(v, w1[t], a1);
      a2 = fmaf(v, w2[t], a2);
    }
  }
  head[((size_t)b * 3 + 0) * HW + p] = 0.25f * tanhf(a0);         // dx
  head[((size_t)b * 3 + 1) * HW + p] = 0.25f * tanhf(a1);         // dy
  head[((size_t)b * 3 + 2) * HW + p] = 1.0f / (1.0f + expf(-a2)); // mask
}

// ------------- bilinear gather + mask -> packed f16 im2col V[n][k], k=c*9+tap -------------
__global__ __launch_bounds__(256) void im2col_kernel(
    const float* __restrict__ src, const float* __restrict__ head,
    _Float16* __restrict__ V)
{
  int tid = blockIdx.x * 256 + threadIdx.x;
  if (tid >= BATCH * HW * 9) return;
  int tap  = tid % 9;
  int rest = tid / 9;
  int p = rest % HW, b = rest / HW;
  int y = p / WW, x = p % WW;
  float dx = head[((size_t)b * 3 + 0) * HW + p];
  float dy = head[((size_t)b * 3 + 1) * HW + p];
  float mk = head[((size_t)b * 3 + 2) * HW + p];
  // serpentine: off_y coefficient on dx, off_x coefficient on dy
  const int SY[9] = {0, 0, 0, 1, 0, -1, 1, 1, -1};
  const int SX[9] = {0, 1, -1, 0, 0, 0, 1, -1, 1};
  float py = (float)(y + tap / 3 - 1) + (float)SY[tap] * dx;
  float px = (float)(x + tap % 3 - 1) + (float)SX[tap] * dy;
  float y0f = floorf(py), x0f = floorf(px);
  float wy = py - y0f, wx = px - x0f;
  int y0 = (int)y0f, x0 = (int)x0f;
  int yc[4] = {y0, y0, y0 + 1, y0 + 1};
  int xc[4] = {x0, x0 + 1, x0, x0 + 1};
  float wt[4] = {(1 - wy) * (1 - wx), (1 - wy) * wx, wy * (1 - wx), wy * wx};
  int idx[4];
  #pragma unroll
  for (int j = 0; j < 4; ++j) {
    bool valid = (yc[j] >= 0) && (yc[j] < HH) && (xc[j] >= 0) && (xc[j] < WW);
    wt[j] *= valid ? mk : 0.0f;
    int yi = min(max(yc[j], 0), HH - 1);
    int xi = min(max(xc[j], 0), WW - 1);
    idx[j] = yi * WW + xi;
  }
  const float* xb = src + (size_t)b * CH * HW;
  _Float16* vout = V + (size_t)(b * HW + p) * KTOT + tap;
  for (int c = 0; c < CH; ++c) {
    const float* xcp = xb + (size_t)c * HW;
    float v = wt[0] * xcp[idx[0]] + wt[1] * xcp[idx[1]]
            + wt[2] * xcp[idx[2]] + wt[3] * xcp[idx[3]];
    vout[c * 9] = (_Float16)v;
  }
}

// ---------------- fp32 -> f16 weight convert (dcn_w flat [Cout, 576]) ----------------
__global__ __launch_bounds__(256) void wconv_kernel(
    const float* __restrict__ ws, _Float16* __restrict__ wh, int n)
{
  int tid = blockIdx.x * 256 + threadIdx.x;
  if (tid < n) wh[tid] = (_Float16)ws[tid];
}

// ---------------- DCN contraction as WMMA GEMM: Y[64,N] = Wh[64,576] x V[576,N] ----------------
// A (weights, 72 KB f16) is staged once per block into LDS via async global->LDS
// copy (ASYNCcnt-tracked), then fed to the WMMAs with ds_load_b128 while the
// streaming im2col B operand comes straight from global (B128 loads + prefetch).
__global__ __launch_bounds__(256) void dcn_gemm_kernel(
    const _Float16* __restrict__ Wh, const _Float16* __restrict__ V,
    float* __restrict__ Y)
{
  __shared__ _Float16 Alds[CH * KTOT];   // 64*576*2B = 72 KB of the WGP's 320 KB

#if HAVE_ASYNC_LDS
  for (int i = threadIdx.x; i < (CH * KTOT) / 8; i += 256) {
    __builtin_amdgcn_global_load_async_to_lds_b128(
        (g_v4i*)(Wh + (size_t)i * 8), (l_v4i*)(&Alds[i * 8]),
        /*offset=*/0, /*cpol=*/0);
  }
  __builtin_amdgcn_s_wait_asynccnt(0);  // wave's own async copies done
#else
  for (int i = threadIdx.x; i < (CH * KTOT) / 8; i += 256) {
    *(v8h*)(&Alds[i * 8]) = *(const v8h*)(Wh + (size_t)i * 8);
  }
#endif
  __syncthreads();

  int wave = threadIdx.x >> 5;
  int lane = threadIdx.x & 31;
  int mtile = wave & 3;   // 4 tiles of 16 output channels
  int ntile = wave >> 2;  // 2 tiles of 16 pixels
  int half  = lane >> 4;
  int nl    = lane & 15;
  size_t ncol = (size_t)blockIdx.x * 32 + (size_t)ntile * 16 + nl;
  const _Float16* Arow = &Alds[(size_t)(mtile * 16 + nl) * KTOT]; // A row M = nl (LDS)
  const _Float16* Brow = V + ncol * KTOT;                         // B col N = nl (global)
  v8f acc = {};
  for (int kb = 0; kb < KTOT; kb += 32) {
    // A 16x32 f16 layout: lanes 0-15 hold K kb+0..7 / kb+16..23; lanes 16-31 hold kb+8..15 / kb+24..31
    union { v16h v; v8h h[2]; } a;
    a.h[0] = *(const v8h*)(Arow + kb + half * 8);
    a.h[1] = *(const v8h*)(Arow + kb + 16 + half * 8);
    // B 32x16 f16: same K split as A: lanes 0-15 hold K kb+0..7 / kb+16..23; lanes 16-31 hold kb+8..15 / kb+24..31
    union { v16h v; v8h h[2]; } bu;
    bu.h[0] = *(const v8h*)(Brow + kb + half * 8);
    bu.h[1] = *(const v8h*)(Brow + kb + 16 + half * 8);
    v16h bm = bu.v;
    if (kb + 32 < KTOT)
      __builtin_prefetch(Brow + kb + 32, 0, 1);   // global_prefetch_b8
    acc = __builtin_amdgcn_wmma_f32_16x16x32_f16(
        /*neg_a=*/false, a.v, /*neg_b=*/false, bm,
        /*c_mod=*/(short)0, acc, /*reuse_a=*/false, /*reuse_b=*/false);
  }
  // D layout: lanes 0-15: VGPR r -> M=r; lanes 16-31: M=r+8; N = lane%16
  int bimg = (int)(ncol >> 14);
  int p    = (int)(ncol & (HW - 1));
  int obase = mtile * 16 + half * 8;
  #pragma unroll
  for (int r = 0; r < 8; ++r)
    Y[((size_t)bimg * CH + obase + r) * HW + p] = acc[r];
}

// ---------------- GroupNorm stats: one block per (b, group) ----------------
__global__ __launch_bounds__(256) void gn_stats_kernel(
    const float* __restrict__ Y, float* __restrict__ stats)
{
  __shared__ float s1[256], s2[256];
  int bg = blockIdx.x;                         // b*8 + g
  const float* base = Y + (size_t)bg * 8 * HW; // contiguous 8-channel group
  float s = 0.f, ss = 0.f;
  for (int i = threadIdx.x; i < 8 * HW; i += 256) {
    float v = base[i];
    s += v; ss += v * v;
  }
  s1[threadIdx.x] = s; s2[threadIdx.x] = ss;
  __syncthreads();
  for (int off = 128; off > 0; off >>= 1) {
    if ((int)threadIdx.x < off) {
      s1[threadIdx.x] += s1[threadIdx.x + off];
      s2[threadIdx.x] += s2[threadIdx.x + off];
    }
    __syncthreads();
  }
  if (threadIdx.x == 0) {
    float n = 8.0f * HW;
    float mu = s1[0] / n;
    float var = s2[0] / n - mu * mu;
    stats[bg * 2 + 0] = mu;
    stats[bg * 2 + 1] = rsqrtf(var + GN_EPS);
  }
}

// ---------------- GroupNorm apply + affine + ReLU ----------------
__global__ __launch_bounds__(256) void gn_apply_kernel(
    const float* __restrict__ Y, const float* __restrict__ stats,
    const float* __restrict__ gam, const float* __restrict__ bet,
    float* __restrict__ dst)
{
  size_t tid = (size_t)blockIdx.x * 256 + threadIdx.x;
  if (tid >= (size_t)BATCH * CH * HW) return;
  int o = (int)((tid >> 14) & (CH - 1));
  int b = (int)(tid >> 20);
  int bg = b * 8 + (o >> 3);
  float mu = stats[bg * 2 + 0];
  float rs = stats[bg * 2 + 1];
  float v = (Y[tid] - mu) * rs * gam[o] + bet[o];
  dst[tid] = v > 0.f ? v : 0.f;
}

// ---------------- per-layer pipeline ----------------
static void run_layer(const float* src, const float* ohw, const float* ohb,
                      const float* dcnw, const float* gng, const float* gnb,
                      float* dst, float* head, _Float16* Wh, _Float16* V,
                      float* Y, float* stats, hipStream_t stream)
{
  head_kernel<<<(BATCH * HW + 255) / 256, 256, 0, stream>>>(src, ohw, ohb, head);
  im2col_kernel<<<(BATCH * HW * 9 + 255) / 256, 256, 0, stream>>>(src, head, V);
  wconv_kernel<<<(CH * KTOT + 255) / 256, 256, 0, stream>>>(dcnw, Wh, CH * KTOT);
  dcn_gemm_kernel<<<NTOT / 32, 256, 0, stream>>>(Wh, V, Y);
  gn_stats_kernel<<<BATCH * 8, 256, 0, stream>>>(Y, stats);
  gn_apply_kernel<<<(BATCH * CH * HW + 255) / 256, 256, 0, stream>>>(Y, stats, gng, gnb, dst);
}

extern "C" void kernel_launch(void* const* d_in, const int* in_sizes, int n_in,
                              void* d_out, int out_size, void* d_ws, size_t ws_size,
                              hipStream_t stream)
{
  const float* x     = (const float*)d_in[0];
  const float* oh1w  = (const float*)d_in[1];
  const float* oh1b  = (const float*)d_in[2];
  const float* dcn1w = (const float*)d_in[3];
  const float* gn1g  = (const float*)d_in[4];
  const float* gn1b  = (const float*)d_in[5];
  const float* oh2w  = (const float*)d_in[6];
  const float* oh2b  = (const float*)d_in[7];
  const float* dcn2w = (const float*)d_in[8];
  const float* gn2g  = (const float*)d_in[9];
  const float* gn2b  = (const float*)d_in[10];
  float* out = (float*)d_out;

  char* ws = (char*)d_ws;
  size_t off = 0;
  auto alloc = [&](size_t bytes) -> char* {
    off = (off + 255) & ~(size_t)255;
    char* p = ws + off;
    off += bytes;
    return p;
  };
  float*    head  = (float*)alloc(sizeof(float) * (size_t)BATCH * 3 * HW);
  _Float16* Wh    = (_Float16*)alloc(sizeof(_Float16) * (size_t)CH * KTOT);
  float*    stats = (float*)alloc(sizeof(float) * BATCH * 8 * 2);
  float*    Y     = (float*)alloc(sizeof(float) * (size_t)BATCH * CH * HW);
  float*    Hbuf  = (float*)alloc(sizeof(float) * (size_t)BATCH * CH * HW);
  _Float16* V     = (_Float16*)alloc(sizeof(_Float16) * (size_t)NTOT * KTOT);
  (void)ws_size; (void)in_sizes; (void)n_in; (void)out_size;

  run_layer(x,    oh1w, oh1b, dcn1w, gn1g, gn1b, Hbuf, head, Wh, V, Y, stats, stream);
  run_layer(Hbuf, oh2w, oh2b, dcn2w, gn2g, gn2b, out,  head, Wh, V, Y, stats, stream);
}
